// ChannelAwareAttentionModule_38422777430174
// MI455X (gfx1250) — matrix operand order repaired and checked by reading, not compiled, not run on hardware
//
#include <hip/hip_runtime.h>

// ---------------------------------------------------------------------------
// CDNA5 (gfx1250) channel-attention module.
// All GEMMs via v_wmma_f32_16x16x32_bf16; TDM used for the bf16 f-matrix tile.
// ---------------------------------------------------------------------------

typedef __attribute__((ext_vector_type(16))) __bf16 v16bf;
typedef __attribute__((ext_vector_type(8)))  __bf16 v8bf;
typedef __attribute__((ext_vector_type(8)))  float  v8f;
typedef unsigned int v4u __attribute__((ext_vector_type(4)));
typedef int          v4i __attribute__((ext_vector_type(4)));
typedef int          v8i __attribute__((ext_vector_type(8)));

#define DEVI __device__ __forceinline__

static constexpr int   kB   = 16;
static constexpr int   kC   = 256;
static constexpr int   kCi  = 128;
static constexpr int   kN   = 4096;   // H*W
static constexpr float kEPS = 1e-5f;

DEVI void cvt4_store(__bf16* dst, float4 v) {   // 4 fp32 -> 4 bf16, 8B LDS store
  union { __bf16 h[4]; uint2 u; } p;
  p.h[0] = (__bf16)v.x; p.h[1] = (__bf16)v.y;
  p.h[2] = (__bf16)v.z; p.h[3] = (__bf16)v.w;
  *reinterpret_cast<uint2*>(dst) = p.u;
}

DEVI void cvt2_store(__bf16* dst, float a, float b) { // pair -> 4B LDS store
  union { __bf16 h[2]; unsigned u; } p;
  p.h[0] = (__bf16)a; p.h[1] = (__bf16)b;
  *reinterpret_cast<unsigned*>(dst) = p.u;
}

DEVI unsigned short f2bf_bits(float f) {
  union { __bf16 h; unsigned short s; } p;
  p.h = (__bf16)f;
  return p.s;
}

// two 16B LDS chunks -> one v16bf WMMA operand fragment (ds_load_b128 x2)
DEVI v16bf ldfrag(const __bf16* p0, const __bf16* p1) {
  v8bf lo = *reinterpret_cast<const v8bf*>(p0);
  v8bf hi = *reinterpret_cast<const v8bf*>(p1);
  v16bf r;
#pragma unroll
  for (int i = 0; i < 8; ++i) { r[i] = lo[i]; r[i + 8] = hi[i]; }
  return r;
}

DEVI v8f zero8() {
  v8f a;
#pragma unroll
  for (int i = 0; i < 8; ++i) a[i] = 0.f;
  return a;
}

DEVI v8f wmma_bf16(v16bf a, v16bf b, v8f c) {
  return __builtin_amdgcn_wmma_f32_16x16x32_bf16(false, a, false, b,
                                                 (short)0, c, false, false);
}

// ---------------------------------------------------------------------------
// Kernel 1: fused 1x1 conv + BN + ReLU.  M=128, K=256, N-tile=128.
// Double-buffered LDS pipeline: global loads for step k+1 overlap WMMA of k.
// grid.x = 16 batches * 32 n-tiles = 512, block = 256 (8 waves)
// ---------------------------------------------------------------------------
DEVI void proj_issue(const float* xb, int k0, int t, float4 (&r)[2][2]) {
#pragma unroll
  for (int j = 0; j < 2; ++j) {
    int it = t + 256 * j;
    int kp = it >> 5;            // k-pair 0..15
    int nq = (it & 31) * 4;      // n quad
    const float* p0 = xb + (size_t)(k0 + 2 * kp) * kN + nq;
    r[j][0] = *reinterpret_cast<const float4*>(p0);
    r[j][1] = *reinterpret_cast<const float4*>(p0 + kN);
  }
}
DEVI void proj_store(__bf16* X, int t, float4 (&r)[2][2]) {
#pragma unroll
  for (int j = 0; j < 2; ++j) {
    int it = t + 256 * j;
    int kp = it >> 5;
    int nq = (it & 31) * 4;
    cvt2_store(&X[(nq + 0) * 40 + 2 * kp], r[j][0].x, r[j][1].x);
    cvt2_store(&X[(nq + 1) * 40 + 2 * kp], r[j][0].y, r[j][1].y);
    cvt2_store(&X[(nq + 2) * 40 + 2 * kp], r[j][0].z, r[j][1].z);
    cvt2_store(&X[(nq + 3) * 40 + 2 * kp], r[j][0].w, r[j][1].w);
  }
}

__global__ __launch_bounds__(256)
void proj3_kernel(const float* __restrict__ x,       // [16][256][4096]
                  const float* __restrict__ w,       // [128][256]
                  const float* __restrict__ bias, const float* __restrict__ gamma,
                  const float* __restrict__ beta, const float* __restrict__ mean,
                  const float* __restrict__ var,
                  float* __restrict__ outbuf)        // [16][128][4096]
{
  __shared__ __attribute__((aligned(16))) __bf16 Wl[128 * 256];   // [o][k]
  __shared__ __attribute__((aligned(16))) __bf16 Xl[2][128 * 40]; // [n][k] pad 40
  __shared__ float scl[128], sft[128];

  const int t    = threadIdx.x;
  const int b    = blockIdx.x >> 5;
  const int n0   = (blockIdx.x & 31) * 128;
  const int wid  = t >> 5;
  const int lane = t & 31;
  const int gsel = lane >> 4;
  const int lm   = lane & 15;

  for (int i = t * 4; i < 128 * 256; i += 256 * 4)
    cvt4_store(&Wl[i], *reinterpret_cast<const float4*>(w + i));
  if (t < 128) {
    float s = gamma[t] * rsqrtf(var[t] + kEPS);
    scl[t] = s;
    sft[t] = (bias[t] - mean[t]) * s + beta[t];
  }

  v8f acc[8];
#pragma unroll
  for (int i = 0; i < 8; ++i) acc[i] = zero8();

  const float* xb = x + (size_t)b * kC * kN + n0;

  float4 regs[2][2];
  proj_issue(xb, 0, t, regs);
  proj_store(Xl[0], t, regs);
  __syncthreads();

  for (int step = 0; step < 8; ++step) {
    const __bf16* Xc = Xl[step & 1];
    const bool more = (step + 1) < 8;
    if (more) proj_issue(xb, 32 * (step + 1), t, regs);   // overlap w/ compute

    const __bf16* wrow = &Wl[(16 * wid + lm) * 256 + 32 * step];
    v16bf afr = ldfrag(wrow + gsel * 8, wrow + 16 + gsel * 8);
#pragma unroll
    for (int h = 0; h < 2; ++h) {
      v16bf bfr[4];
#pragma unroll
      for (int j = 0; j < 4; ++j) {
        const __bf16* xrow = &Xc[((h * 4 + j) * 16 + lm) * 40 + gsel * 8];
        bfr[j] = ldfrag(xrow, xrow + 16);
      }
#pragma unroll
      for (int j = 0; j < 4; ++j)
        acc[h * 4 + j] = wmma_bf16(afr, bfr[j], acc[h * 4 + j]);
    }
    if (more) proj_store(Xl[(step & 1) ^ 1], t, regs);
    __syncthreads();
  }

  float* ob = outbuf + (size_t)b * kCi * kN + n0;
#pragma unroll
  for (int v = 0; v < 8; ++v) {
    int o = 16 * wid + gsel * 8 + v;
    float s = scl[o], sh = sft[o];
#pragma unroll
    for (int ns = 0; ns < 8; ++ns)
      ob[(size_t)o * kN + ns * 16 + lm] = fmaxf(acc[ns][v] * s + sh, 0.f);
  }
}

// ---------------------------------------------------------------------------
// Kernel 2a: S_partial = theta * phi^T, K-split x8, double-buffered.
// grid (16 batches, 8 splits)
// ---------------------------------------------------------------------------
DEVI void score_issue(const float* tb, const float* pb, int k0, int t,
                      float4 (&r)[8]) {
#pragma unroll
  for (int j = 0; j < 8; ++j) {
    int it = t + 256 * j;
    int which = it >> 10;
    int row   = (it >> 3) & 127;
    int q     = (it & 7) * 4;
    const float* src = (which ? pb : tb) + (size_t)row * kN + k0 + q;
    r[j] = *reinterpret_cast<const float4*>(src);
  }
}
DEVI void score_store(__bf16* T, __bf16* P, int t, float4 (&r)[8]) {
#pragma unroll
  for (int j = 0; j < 8; ++j) {
    int it = t + 256 * j;
    int which = it >> 10;
    int row   = (it >> 3) & 127;
    int q     = (it & 7) * 4;
    cvt4_store((which ? P : T) + row * 40 + q, r[j]);
  }
}

__global__ __launch_bounds__(256)
void score_kernel(const float* __restrict__ th, const float* __restrict__ ph,
                  float* __restrict__ spart)   // [16][8][128][128]
{
  __shared__ __attribute__((aligned(16))) __bf16 Tl[2][128 * 40];  // [c][k]
  __shared__ __attribute__((aligned(16))) __bf16 Pl[2][128 * 40];  // [d][k]

  const int t    = threadIdx.x;
  const int b    = blockIdx.x;
  const int ks   = blockIdx.y;
  const int wid  = t >> 5;
  const int lane = t & 31;
  const int gsel = lane >> 4;
  const int lm   = lane & 15;

  const float* tb = th + (size_t)b * kCi * kN;
  const float* pb = ph + (size_t)b * kCi * kN;

  v8f acc[8];
#pragma unroll
  for (int i = 0; i < 8; ++i) acc[i] = zero8();

  float4 regs[8];
  score_issue(tb, pb, ks * 512, t, regs);
  score_store(Tl[0], Pl[0], t, regs);
  __syncthreads();

  for (int step = 0; step < 16; ++step) {
    const __bf16* Tc = Tl[step & 1];
    const __bf16* Pc = Pl[step & 1];
    const bool more = (step + 1) < 16;
    if (more) score_issue(tb, pb, ks * 512 + 32 * (step + 1), t, regs);

    const __bf16* trow = &Tc[(16 * wid + lm) * 40];
    v16bf afr = ldfrag(trow + gsel * 8, trow + 16 + gsel * 8);
#pragma unroll
    for (int h = 0; h < 2; ++h) {
      v16bf bfr[4];
#pragma unroll
      for (int j = 0; j < 4; ++j) {
        const __bf16* prow = &Pc[((h * 4 + j) * 16 + lm) * 40 + gsel * 8];
        bfr[j] = ldfrag(prow, prow + 16);
      }
#pragma unroll
      for (int j = 0; j < 4; ++j)
        acc[h * 4 + j] = wmma_bf16(afr, bfr[j], acc[h * 4 + j]);
    }
    if (more) score_store(Tl[(step & 1) ^ 1], Pl[(step & 1) ^ 1], t, regs);
    __syncthreads();
  }

  float* sp = spart + ((size_t)(b * 8 + ks) * 128) * 128;
#pragma unroll
  for (int v = 0; v < 8; ++v) {
    int c = 16 * wid + gsel * 8 + v;
#pragma unroll
    for (int ns = 0; ns < 8; ++ns)
      sp[(size_t)c * 128 + ns * 16 + lm] = acc[ns][v];
  }
}

// ---------------------------------------------------------------------------
// Kernel 2b: reduce K-split partials + row softmax -> bf16 f.  wave per row.
// ---------------------------------------------------------------------------
__global__ __launch_bounds__(256)
void softmax_kernel(const float* __restrict__ spart,
                    unsigned short* __restrict__ fbuf) // bf16 [16][128][128]
{
  const int row  = blockIdx.x * 8 + (threadIdx.x >> 5);
  const int b    = row >> 7;
  const int c    = row & 127;
  const int lane = threadIdx.x & 31;

  float vals[4];
#pragma unroll
  for (int j = 0; j < 4; ++j) {
    int d = lane + 32 * j;
    float s = 0.f;
#pragma unroll
    for (int p = 0; p < 8; ++p)
      s += spart[(((size_t)b * 8 + p) * 128 + c) * 128 + d];
    vals[j] = s;
  }
  float mx = fmaxf(fmaxf(vals[0], vals[1]), fmaxf(vals[2], vals[3]));
#pragma unroll
  for (int off = 16; off > 0; off >>= 1) mx = fmaxf(mx, __shfl_xor(mx, off, 32));
  float sum = 0.f;
#pragma unroll
  for (int j = 0; j < 4; ++j) { vals[j] = __expf(vals[j] - mx); sum += vals[j]; }
#pragma unroll
  for (int off = 16; off > 0; off >>= 1) sum += __shfl_xor(sum, off, 32);
  float inv = 1.f / sum;
#pragma unroll
  for (int j = 0; j < 4; ++j)
    fbuf[((size_t)b * 128 + c) * 128 + lane + 32 * j] = f2bf_bits(vals[j] * inv);
}

// ---------------------------------------------------------------------------
// Kernel 2c: y = f * g, stored transposed (permute/view quirk == y^T):
//            z[b][n][c] = y[b][c][n].   grid.x = 512 (batch, n-tile)
// f tile staged via the Tensor Data Mover (bf16 -> LDS raw copy w/ row pad).
// ---------------------------------------------------------------------------
__global__ __launch_bounds__(256)
void apply_kernel(const unsigned short* __restrict__ fbuf, // bf16 [16][128][128]
                  const float* __restrict__ g,             // [16][128][4096]
                  float* __restrict__ z)                   // [16][4096][128]
{
  __shared__ __attribute__((aligned(16))) __bf16 Fl[128 * 136]; // [c][d] pad 136
  __shared__ __attribute__((aligned(16))) __bf16 Gl[128 * 136]; // [n][d] pad 136

  const int t    = threadIdx.x;
  const int b    = blockIdx.x >> 5;
  const int n0   = (blockIdx.x & 31) * 128;
  const int wid  = t >> 5;
  const int lane = t & 31;
  const int gsel = lane >> 4;
  const int lm   = lane & 15;

  // ---- TDM: DMA the 128x128 bf16 f-matrix into Fl with 16B row padding ----
  if (wid == 0) {
    unsigned long long ga =
        (unsigned long long)(uintptr_t)(fbuf + (size_t)b * 128 * 128);
    unsigned lds = (unsigned)(uintptr_t)(&Fl[0]);
    v4u g0;
    g0[0] = 1u;                                   // count=1, user descriptor
    g0[1] = lds;                                  // lds_addr
    g0[2] = (unsigned)ga;                         // global_addr[31:0]
    g0[3] = (unsigned)((ga >> 32) & 0x1FFFFFFu) | (2u << 30); // addr hi | type=2
    v8i g1;
    g1[0] = (int)((1u << 16) |                    // data_size = 2 bytes
                  (1u << 20) |                    // pad_enable
                  (5u << 22) |                    // pad_interval: 64 DWORDs (row)
                  (3u << 25));                    // pad_amount: 4 DWORDs (16B)
    g1[1] = (int)(128u << 16);                    // tensor_dim0 = 128
    g1[2] = (int)(128u << 16);                    // tensor_dim1 = 128
    g1[3] = (int)(128u << 16);                    // tile_dim0 = 128
    g1[4] = 128;                                  // tile_dim1 = 128
    g1[5] = 128;                                  // tensor_dim0_stride = 128
    g1[6] = 0;
    g1[7] = 0;
    v4i g2 = {0, 0, 0, 0}, g3 = {0, 0, 0, 0};
    v8i g4 = {0, 0, 0, 0, 0, 0, 0, 0};
    __builtin_amdgcn_tensor_load_to_lds(g0, g1, g2, g3, g4, 0);
  }

  // stage g transposed: Gl[n][d]
  const float* gb = g + (size_t)b * kCi * kN + n0;
  for (int it = t; it < 2048; it += 256) {   // 64 d-pairs * 32 n-quads
    int dp = it >> 5;
    int nq = (it & 31) * 4;
    const float* p0 = gb + (size_t)(2 * dp) * kN + nq;
    const float* p1 = p0 + kN;
    float4 a = *reinterpret_cast<const float4*>(p0);
    float4 c = *reinterpret_cast<const float4*>(p1);
    cvt2_store(&Gl[(nq + 0) * 136 + 2 * dp], a.x, c.x);
    cvt2_store(&Gl[(nq + 1) * 136 + 2 * dp], a.y, c.y);
    cvt2_store(&Gl[(nq + 2) * 136 + 2 * dp], a.z, c.z);
    cvt2_store(&Gl[(nq + 3) * 136 + 2 * dp], a.w, c.w);
  }
  __builtin_amdgcn_s_wait_tensorcnt(0);
  __syncthreads();

  v8f acc[8];
#pragma unroll
  for (int i = 0; i < 8; ++i) acc[i] = zero8();

#pragma unroll
  for (int k0 = 0; k0 < 128; k0 += 32) {
    const __bf16* frow = &Fl[(16 * wid + lm) * 136 + k0];
    v16bf afr = ldfrag(frow + gsel * 8, frow + 16 + gsel * 8);
#pragma unroll
    for (int h = 0; h < 2; ++h) {
      v16bf bfr[4];
#pragma unroll
      for (int j = 0; j < 4; ++j) {
        const __bf16* grow = &Gl[((h * 4 + j) * 16 + lm) * 136 + k0 + gsel * 8];
        bfr[j] = ldfrag(grow, grow + 16);
      }
#pragma unroll
      for (int j = 0; j < 4; ++j)
        acc[h * 4 + j] = wmma_bf16(afr, bfr[j], acc[h * 4 + j]);
    }
  }

  float* zb = z + (size_t)b * kN * kCi;
#pragma unroll
  for (int v = 0; v < 8; ++v) {
    int c = 16 * wid + gsel * 8 + v;
#pragma unroll
    for (int ns = 0; ns < 8; ++ns)
      zb[(size_t)(n0 + ns * 16 + lm) * kCi + c] = acc[ns][v];
  }
}

// ---------------------------------------------------------------------------
// Kernel 3: out = BN(W2 * z^T) + x.  M=256, K=128, N-tile=128. grid.x = 512
// z layout [b][n][k] means B-fragments read contiguous K. No ReLU.
// ---------------------------------------------------------------------------
__global__ __launch_bounds__(256)
void final_kernel(const float* __restrict__ z,    // [16][4096][128]
                  const float* __restrict__ w2,   // [256][128]
                  const float* __restrict__ bias, const float* __restrict__ gamma,
                  const float* __restrict__ beta, const float* __restrict__ mean,
                  const float* __restrict__ var,
                  const float* __restrict__ x,    // [16][256][4096]
                  float* __restrict__ out)        // [16][256][4096]
{
  __shared__ __attribute__((aligned(16))) __bf16 W2l[256 * 128]; // [o][k]
  __shared__ __attribute__((aligned(16))) __bf16 Zl[128 * 136];  // [n][k]
  __shared__ float scl[256], sft[256];

  const int t    = threadIdx.x;
  const int b    = blockIdx.x >> 5;
  const int n0   = (blockIdx.x & 31) * 128;
  const int wid  = t >> 5;
  const int lane = t & 31;
  const int gsel = lane >> 4;
  const int lm   = lane & 15;

  for (int i = t * 4; i < 256 * 128; i += 256 * 4)
    cvt4_store(&W2l[i], *reinterpret_cast<const float4*>(w2 + i));
  if (t < 256) {
    float s = gamma[t] * rsqrtf(var[t] + kEPS);
    scl[t] = s;
    sft[t] = (bias[t] - mean[t]) * s + beta[t];
  }
  const float* zb = z + ((size_t)b * kN + n0) * kCi;
  for (int it = t; it < 4096; it += 256) {     // 128 rows * 32 quads
    int row = it >> 5, q = (it & 31) * 4;
    cvt4_store(&Zl[row * 136 + q],
               *reinterpret_cast<const float4*>(zb + (size_t)row * kCi + q));
  }
  __syncthreads();

  v8f acc[16];   // 2 M-strips x 8 N-subtiles
#pragma unroll
  for (int i = 0; i < 16; ++i) acc[i] = zero8();

#pragma unroll
  for (int k0 = 0; k0 < 128; k0 += 32) {
    v16bf afr[2];
#pragma unroll
    for (int ms = 0; ms < 2; ++ms) {
      const __bf16* wrow = &W2l[(128 * ms + 16 * wid + lm) * 128 + k0];
      afr[ms] = ldfrag(wrow + gsel * 8, wrow + 16 + gsel * 8);
    }
#pragma unroll
    for (int h = 0; h < 2; ++h) {
      v16bf bfr[4];
#pragma unroll
      for (int j = 0; j < 4; ++j) {
        const __bf16* zrow = &Zl[((h * 4 + j) * 16 + lm) * 136 + k0 + gsel * 8];
        bfr[j] = ldfrag(zrow, zrow + 16);
      }
#pragma unroll
      for (int j = 0; j < 4; ++j) {
        acc[h * 4 + j]     = wmma_bf16(afr[0], bfr[j], acc[h * 4 + j]);
        acc[8 + h * 4 + j] = wmma_bf16(afr[1], bfr[j], acc[8 + h * 4 + j]);
      }
    }
  }

  const float* xb = x + (size_t)b * kC * kN + n0;
  float* ob = out + (size_t)b * kC * kN + n0;
#pragma unroll
  for (int ms = 0; ms < 2; ++ms) {
#pragma unroll
    for (int v = 0; v < 8; ++v) {
      int o = 128 * ms + 16 * wid + gsel * 8 + v;
      float s = scl[o], sh = sft[o];
#pragma unroll
      for (int ns = 0; ns < 8; ++ns) {
        size_t idx = (size_t)o * kN + ns * 16 + lm;
        ob[idx] = acc[ms * 8 + ns][v] * s + sh + xb[idx];
      }
    }
  }
}

// ---------------------------------------------------------------------------
extern "C" void kernel_launch(void* const* d_in, const int* in_sizes, int n_in,
                              void* d_out, int out_size, void* d_ws, size_t ws_size,
                              hipStream_t stream)
{
  const float* x   = (const float*)d_in[0];
  const float* gw  = (const float*)d_in[1];
  const float* gb  = (const float*)d_in[2];
  const float* gg  = (const float*)d_in[3];
  const float* gbe = (const float*)d_in[4];
  const float* gm  = (const float*)d_in[5];
  const float* gv  = (const float*)d_in[6];
  const float* tw  = (const float*)d_in[7];
  const float* tb  = (const float*)d_in[8];
  const float* tg  = (const float*)d_in[9];
  const float* tbe = (const float*)d_in[10];
  const float* tm  = (const float*)d_in[11];
  const float* tv  = (const float*)d_in[12];
  const float* pw  = (const float*)d_in[13];
  const float* pb  = (const float*)d_in[14];
  const float* pg  = (const float*)d_in[15];
  const float* pbe = (const float*)d_in[16];
  const float* pm  = (const float*)d_in[17];
  const float* pv  = (const float*)d_in[18];
  const float* ww  = (const float*)d_in[19];
  const float* wb  = (const float*)d_in[20];
  const float* wg  = (const float*)d_in[21];
  const float* wbe = (const float*)d_in[22];
  const float* wm  = (const float*)d_in[23];
  const float* wv  = (const float*)d_in[24];

  char* ws = (char*)d_ws;
  const size_t SZ = (size_t)kB * kCi * kN * sizeof(float);   // 32 MB
  float*          gbuf  = (float*)(ws);
  float*          thbuf = (float*)(ws + SZ);
  float*          phbuf = (float*)(ws + 2 * SZ);
  float*          zbuf  = (float*)(ws + 3 * SZ);
  float*          spart = (float*)(ws + 4 * SZ);             // 8 MB
  unsigned short* fbuf  = (unsigned short*)(ws + 4 * SZ + (size_t)kB * 8 * 128 * 128 * 4);

  dim3 blk(256);
  proj3_kernel<<<512, blk, 0, stream>>>(x, gw, gb, gg, gbe, gm, gv, gbuf);
  proj3_kernel<<<512, blk, 0, stream>>>(x, tw, tb, tg, tbe, tm, tv, thbuf);
  proj3_kernel<<<512, blk, 0, stream>>>(x, pw, pb, pg, pbe, pm, pv, phbuf);
  score_kernel<<<dim3(16, 8), blk, 0, stream>>>(thbuf, phbuf, spart);
  softmax_kernel<<<256, blk, 0, stream>>>(spart, fbuf);
  apply_kernel<<<512, blk, 0, stream>>>(fbuf, gbuf, zbuf);
  final_kernel<<<512, blk, 0, stream>>>(zbuf, ww, wb, wg, wbe, wm, wv, x, (float*)d_out);
}
